// ResidualMultiheadAttention_26508538151395
// MI455X (gfx1250) — matrix operand order repaired and checked by reading, not compiled, not run on hardware
//
#include <hip/hip_runtime.h>
#include <hip/hip_bf16.h>

typedef __attribute__((ext_vector_type(16))) _Float16 v16h;
typedef __attribute__((ext_vector_type(8)))  _Float16 v8h;
typedef __attribute__((ext_vector_type(4)))  _Float16 v4h;
typedef __attribute__((ext_vector_type(8)))  float    v8f;
typedef __attribute__((ext_vector_type(4)))  float    v4f;

#define H16 _Float16
#define NEG_INF (-__builtin_huge_valf())

static constexpr int E  = 1024;
static constexpr int NH = 16;
static constexpr int D  = 64;
static constexpr int L  = 2048;
static constexpr int NB = 2;
static constexpr int S  = 2048;
static constexpr int BH = NB * NH;   // 32

// ---- gfx1250 async global->LDS copy (guarded; falls back to sync loads) ----
#if defined(__has_builtin)
#if __has_builtin(__builtin_amdgcn_global_load_async_to_lds_b128)
#define HAVE_ASYNC_LDS 1
#endif
#if __has_builtin(__builtin_amdgcn_s_wait_asynccnt)
#define WAIT_ASYNC() __builtin_amdgcn_s_wait_asynccnt(0)
#endif
#endif
#ifndef WAIT_ASYNC
#define WAIT_ASYNC() asm volatile("s_wait_asynccnt 0x0" ::: "memory")
#endif

#if defined(HAVE_ASYNC_LDS)
// per compile diagnostic: param0 is 'int __vector(4) __device__ *' (global int4*)
typedef int v4i __attribute__((__vector_size__(16)));
typedef __attribute__((address_space(1))) v4i gv4i;
typedef __attribute__((address_space(3))) v4i lv4i;
#define ASYNC_CP16(gsrc, ldst) \
    __builtin_amdgcn_global_load_async_to_lds_b128((gv4i*)(void*)(gsrc), (lv4i*)(ldst), 0, 0)
#endif

// ---------------------------------------------------------------------------
// f32 -> f16 weight conversion (4 elems / thread)
// ---------------------------------------------------------------------------
__global__ __launch_bounds__(256) void cvt_f32_f16(const float* __restrict__ src,
                                                   _Float16* __restrict__ dst, int n) {
    int i = (blockIdx.x * 256 + threadIdx.x) * 4;
    if (i + 3 < n) {
        v4f x = *(const v4f*)(src + i);
        v4h y;
        y.x = (H16)x.x; y.y = (H16)x.y; y.z = (H16)x.z; y.w = (H16)x.w;
        *(v4h*)(dst + i) = y;
    }
}

// ---------------------------------------------------------------------------
// Projection GEMM: C[m][e] = X[m][:] @ W[:][e], f32 in, f16 out, plain layout.
// Block 256 thr (8 waves). Tile M=128, N=64, Kstep=32. Wave: 16x64 outputs.
// ---------------------------------------------------------------------------
__global__ __launch_bounds__(256)
void proj_gemm(const float* __restrict__ X,      // (4096, 1024) f32
               const _Float16* __restrict__ Wh,  // (1024, 1024) f16 row-major
               _Float16* __restrict__ dst)       // (4096, 1024) f16
{
    __shared__ _Float16 sA[128 * 40];  // row-major, stride 40
    __shared__ _Float16 sB[64 * 40];   // col-major (n, k), stride 40
    const int tid  = threadIdx.x;
    const int lane = tid & 31, wave = tid >> 5;
    const int hl   = lane >> 4, l16 = lane & 15;
    const int m0   = blockIdx.x * 128;
    const int n0   = blockIdx.y * 64;

    v8f acc[4] = {};
    for (int k0 = 0; k0 < 1024; k0 += 32) {
        __syncthreads();
        {   // stage A: 128x32 f32->f16, 16 elems/thread
            int r = tid >> 1, cb = (tid & 1) * 16;
            const float* src = X + (size_t)(m0 + r) * 1024 + k0 + cb;
            _Float16* d = sA + r * 40 + cb;
            #pragma unroll
            for (int j = 0; j < 16; j += 4) {
                v4f x = *(const v4f*)(src + j);
                v4h y;
                y.x = (H16)x.x; y.y = (H16)x.y; y.z = (H16)x.z; y.w = (H16)x.w;
                *(v4h*)(d + j) = y;
            }
        }
        {   // stage B col-major: 32x64 f16, 8 elems/thread (coalesced read)
            int nb = (tid & 7) * 8, k = tid >> 3;
            v8h x = *(const v8h*)(Wh + (size_t)(k0 + k) * 1024 + n0 + nb);
            #pragma unroll
            for (int j = 0; j < 8; ++j) sB[(nb + j) * 40 + k] = x[j];
        }
        __syncthreads();

        v16h a;  // A-frag: lane<16 rows l16 K{0-7,16-23}; lane>=16 K{8-15,24-31}
        {
            const _Float16* p = sA + (wave * 16 + l16) * 40 + hl * 8;
            *(v8h*)&a       = *(const v8h*)p;
            *((v8h*)&a + 1) = *(const v8h*)(p + 16);
        }
        v16h bf[4];  // pipeline: all B-frags first, then WMMA burst
        #pragma unroll
        for (int j = 0; j < 4; ++j) {
            const _Float16* p = sB + (j * 16 + l16) * 40 + hl * 8;
            *(v8h*)&bf[j]       = *(const v8h*)p;
            *((v8h*)&bf[j] + 1) = *(const v8h*)(p + 16);
        }
        #pragma unroll
        for (int j = 0; j < 4; ++j)
            acc[j] = __builtin_amdgcn_wmma_f32_16x16x32_f16(false, a, false, bf[j],
                                                            (short)0, acc[j], false, false);
    }
    // plain epilogue: base + immediate offsets
    _Float16* db = dst + (size_t)(m0 + wave * 16 + hl * 8) * 1024 + n0 + l16;
    #pragma unroll
    for (int j = 0; j < 4; ++j)
        #pragma unroll
        for (int r = 0; r < 8; ++r)
            db[r * 1024 + j * 16] = (H16)acc[j][r];
}

// ---------------------------------------------------------------------------
// Flash attention. q/k/v in plain (t*NB+n, h*64+d) f16 layout; head columns
// gathered on the fly. One 16-row tile per wave, 8 waves/block, S chunks of 64.
// ---------------------------------------------------------------------------
__global__ __launch_bounds__(256)
void attn_kernel(const _Float16* __restrict__ qpr,  // (4096, 1024)
                 const _Float16* __restrict__ kpr,  // (4096, 1024)
                 const _Float16* __restrict__ vpr,  // (4096, 1024)
                 const float* __restrict__ prev,    // (BH, L, S)
                 const unsigned char* __restrict__ amask,  // (L, S)
                 const unsigned char* __restrict__ kpm,    // (NB, S)
                 float* __restrict__ logits_out,    // (BH, L, S)
                 _Float16* __restrict__ opr)        // (4096, 1024) attn out heads
{
    __shared__ _Float16 sK[64 * 72];       // row-major (s, d), stride 72
    __shared__ _Float16 sV[64 * 72];       // col-major (d, s), stride 72
    __shared__ _Float16 sP[8][16 * 72];    // per-wave P tile (m, s), stride 72
    const int tid  = threadIdx.x;
    const int lane = tid & 31, wave = tid >> 5;
    const int hl   = lane >> 4, l16 = lane & 15;
    const int b    = blockIdx.y;
    const int nidx = b >> 4, h = b & 15;   // b = n*NH + h
    const int t0   = blockIdx.x * 128 + wave * 16;
    const float scale = 0.125f;            // D^-0.5

    // Q fragments (16 rows x 64 depth): two A-frags, K halves 0-31, 32-63
    v16h qa[2];
    {
        const _Float16* qp = qpr + ((size_t)(t0 + l16) * NB + nidx) * 1024 + h * 64;
        #pragma unroll
        for (int c = 0; c < 2; ++c) {
            const _Float16* p = qp + c * 32 + hl * 8;
            *(v8h*)&qa[c]       = *(const v8h*)p;
            *((v8h*)&qa[c] + 1) = *(const v8h*)(p + 16);
        }
    }

    // hot-loop base pointers (per-lane, computed once)
    const float*         prevR = prev       + ((size_t)b * L + t0 + hl * 8) * S + l16;
    float*               logR  = logits_out + ((size_t)b * L + t0 + hl * 8) * S + l16;
    const unsigned char* amR   = amask      + (size_t)(t0 + hl * 8) * S + l16;

    float mrow[8], lrow[8];
    v8f o[4] = {};
    #pragma unroll
    for (int r = 0; r < 8; ++r) { mrow[r] = NEG_INF; lrow[r] = 0.f; }

    for (int s0 = 0; s0 < S; s0 += 64) {
        __syncthreads();
        {   // stage K (async DMA to LDS if available) and V (transposed)
            int s = tid >> 2, cb = (tid & 3) * 16;
            const _Float16* ksrc = kpr + ((size_t)(s0 + s) * NB + nidx) * 1024 + h * 64 + cb;
#if defined(HAVE_ASYNC_LDS)
            ASYNC_CP16(ksrc,     sK + s * 72 + cb);
            ASYNC_CP16(ksrc + 8, sK + s * 72 + cb + 8);
#else
            *(v8h*)(sK + s * 72 + cb)     = *(const v8h*)ksrc;
            *(v8h*)(sK + s * 72 + cb + 8) = *(const v8h*)(ksrc + 8);
#endif
            const _Float16* vsrc = vpr + ((size_t)(s0 + s) * NB + nidx) * 1024 + h * 64 + cb;
            v8h v0 = *(const v8h*)vsrc, v1 = *(const v8h*)(vsrc + 8);
            #pragma unroll
            for (int j = 0; j < 8; ++j) {
                sV[(cb + j)     * 72 + s] = v0[j];
                sV[(cb + 8 + j) * 72 + s] = v1[j];
            }
#if defined(HAVE_ASYNC_LDS)
            WAIT_ASYNC();
#endif
        }
        __syncthreads();

        // S = Q.K^T : load all 8 B-frags, then WMMA burst
        v16h kf[8];
        #pragma unroll
        for (int j = 0; j < 4; ++j)
            #pragma unroll
            for (int c = 0; c < 2; ++c) {
                const _Float16* p = sK + (j * 16 + l16) * 72 + c * 32 + hl * 8;
                *(v8h*)&kf[j * 2 + c]       = *(const v8h*)p;
                *((v8h*)&kf[j * 2 + c] + 1) = *(const v8h*)(p + 16);
            }
        v8f sacc[4];
        #pragma unroll
        for (int j = 0; j < 4; ++j) {
            v8f z = {};
            z = __builtin_amdgcn_wmma_f32_16x16x32_f16(false, qa[0], false, kf[j * 2],
                                                       (short)0, z, false, false);
            z = __builtin_amdgcn_wmma_f32_16x16x32_f16(false, qa[1], false, kf[j * 2 + 1],
                                                       (short)0, z, false, false);
            sacc[j] = z;
        }

        // logits = s*scale + prev (+masks), stream out, track chunk max
        unsigned char kp[4];
        #pragma unroll
        for (int j = 0; j < 4; ++j) kp[j] = kpm[(size_t)nidx * S + s0 + j * 16 + l16];

        float cmax[8];
        #pragma unroll
        for (int r = 0; r < 8; ++r) cmax[r] = NEG_INF;
        #pragma unroll
        for (int j = 0; j < 4; ++j) {
            #pragma unroll
            for (int r = 0; r < 8; ++r) {
                float lg = sacc[j][r] * scale + prevR[r * 2048 + s0 + j * 16];
                if (amR[r * 2048 + s0 + j * 16] | kp[j]) lg = NEG_INF;
                logR[r * 2048 + s0 + j * 16] = lg;
                sacc[j][r] = lg;
                cmax[r] = fmaxf(cmax[r], lg);
            }
        }
        // row max across the 16-lane s groups (lanes 0-15: rows r; 16-31: r+8)
        #pragma unroll
        for (int r = 0; r < 8; ++r) {
            float v = cmax[r];
            #pragma unroll
            for (int w = 1; w < 16; w <<= 1) v = fmaxf(v, __shfl_xor(v, w, 32));
            cmax[r] = v;
        }
        // online softmax update
        float alpha[8], mnew[8], csum[8];
        #pragma unroll
        for (int r = 0; r < 8; ++r) {
            mnew[r]  = fmaxf(mrow[r], cmax[r]);
            alpha[r] = (mnew[r] == NEG_INF) ? 0.f : __expf(mrow[r] - mnew[r]);
            mrow[r]  = mnew[r];
            csum[r]  = 0.f;
        }
        #pragma unroll
        for (int j = 0; j < 4; ++j) {
            #pragma unroll
            for (int r = 0; r < 8; ++r) {
                float p = (mnew[r] == NEG_INF) ? 0.f : __expf(sacc[j][r] - mnew[r]);
                csum[r] += p;
                sP[wave][(hl * 8 + r) * 72 + j * 16 + l16] = (H16)p;
            }
        }
        #pragma unroll
        for (int r = 0; r < 8; ++r) {
            float v = csum[r];
            #pragma unroll
            for (int w = 1; w < 16; w <<= 1) v += __shfl_xor(v, w, 32);
            lrow[r] = lrow[r] * alpha[r] + v;
        }
        #pragma unroll
        for (int j = 0; j < 4; ++j)
            #pragma unroll
            for (int r = 0; r < 8; ++r) o[j][r] *= alpha[r];

        __syncthreads();  // make wave's P LDS tile visible cross-lane

        // O += P.V : pipeline A + 8 B-frags, then WMMA burst
        v16h pa[2];
        #pragma unroll
        for (int c = 0; c < 2; ++c) {
            const _Float16* p = &sP[wave][l16 * 72 + c * 32 + hl * 8];
            *(v8h*)&pa[c]       = *(const v8h*)p;
            *((v8h*)&pa[c] + 1) = *(const v8h*)(p + 16);
        }
        v16h vf[8];
        #pragma unroll
        for (int j = 0; j < 4; ++j)
            #pragma unroll
            for (int c = 0; c < 2; ++c) {
                const _Float16* p = sV + (j * 16 + l16) * 72 + c * 32 + hl * 8;
                *(v8h*)&vf[j * 2 + c]       = *(const v8h*)p;
                *((v8h*)&vf[j * 2 + c] + 1) = *(const v8h*)(p + 16);
            }
        #pragma unroll
        for (int j = 0; j < 4; ++j) {
            o[j] = __builtin_amdgcn_wmma_f32_16x16x32_f16(false, pa[0], false, vf[j * 2],
                                                          (short)0, o[j], false, false);
            o[j] = __builtin_amdgcn_wmma_f32_16x16x32_f16(false, pa[1], false, vf[j * 2 + 1],
                                                          (short)0, o[j], false, false);
        }
    }

    // normalize and store head outputs into plain layout (row t*NB+n, col h*64+d)
    float inv[8];
    #pragma unroll
    for (int r = 0; r < 8; ++r) inv[r] = (lrow[r] > 0.f) ? 1.f / lrow[r] : 0.f;
    _Float16* ob = opr + ((size_t)(t0 + hl * 8) * NB + nidx) * 1024 + h * 64 + l16;
    #pragma unroll
    for (int j = 0; j < 4; ++j)
        #pragma unroll
        for (int r = 0; r < 8; ++r)
            ob[r * 2048 + j * 16] = (H16)(o[j][r] * inv[r]);
}

// ---------------------------------------------------------------------------
// Output projection: out[m][e] = opr[m][:] @ Wo[:][e], f16 in, f32 out.
// ---------------------------------------------------------------------------
__global__ __launch_bounds__(256)
void out_gemm(const _Float16* __restrict__ opr,  // (4096, 1024) f16
              const _Float16* __restrict__ Wo,   // (1024, 1024) f16
              float* __restrict__ out)           // (4096, 1024) f32 == (L, NB, E)
{
    __shared__ _Float16 sA[128 * 40];
    __shared__ _Float16 sB[64 * 40];
    const int tid  = threadIdx.x;
    const int lane = tid & 31, wave = tid >> 5;
    const int hl   = lane >> 4, l16 = lane & 15;
    const int m0   = blockIdx.x * 128;
    const int n0   = blockIdx.y * 64;

    v8f acc[4] = {};
    for (int k0 = 0; k0 < 1024; k0 += 32) {
        __syncthreads();
        {   // stage A rows (f16 copy)
            int r = tid >> 1, cb = (tid & 1) * 16;
            const _Float16* src = opr + (size_t)(m0 + r) * 1024 + k0 + cb;
            *(v8h*)(sA + r * 40 + cb)     = *(const v8h*)src;
            *(v8h*)(sA + r * 40 + cb + 8) = *(const v8h*)(src + 8);
        }
        {   // stage B col-major
            int nb = (tid & 7) * 8, k = tid >> 3;
            v8h x = *(const v8h*)(Wo + (size_t)(k0 + k) * 1024 + n0 + nb);
            #pragma unroll
            for (int j = 0; j < 8; ++j) sB[(nb + j) * 40 + k] = x[j];
        }
        __syncthreads();

        v16h a;
        {
            const _Float16* p = sA + (wave * 16 + l16) * 40 + hl * 8;
            *(v8h*)&a       = *(const v8h*)p;
            *((v8h*)&a + 1) = *(const v8h*)(p + 16);
        }
        v16h bf[4];
        #pragma unroll
        for (int j = 0; j < 4; ++j) {
            const _Float16* p = sB + (j * 16 + l16) * 40 + hl * 8;
            *(v8h*)&bf[j]       = *(const v8h*)p;
            *((v8h*)&bf[j] + 1) = *(const v8h*)(p + 16);
        }
        #pragma unroll
        for (int j = 0; j < 4; ++j)
            acc[j] = __builtin_amdgcn_wmma_f32_16x16x32_f16(false, a, false, bf[j],
                                                            (short)0, acc[j], false, false);
    }
    float* ob = out + (size_t)(m0 + wave * 16 + hl * 8) * 1024 + n0 + l16;
    #pragma unroll
    for (int j = 0; j < 4; ++j)
        #pragma unroll
        for (int r = 0; r < 8; ++r)
            ob[r * 1024 + j * 16] = acc[j][r];
}

// ---------------------------------------------------------------------------
extern "C" void kernel_launch(void* const* d_in, const int* in_sizes, int n_in,
                              void* d_out, int out_size, void* d_ws, size_t ws_size,
                              hipStream_t stream) {
    (void)in_sizes; (void)n_in; (void)out_size; (void)ws_size;
    const float* query = (const float*)d_in[0];
    const float* key   = (const float*)d_in[1];
    const float* value = (const float*)d_in[2];
    const float* prev  = (const float*)d_in[3];
    const unsigned char* kpm   = (const unsigned char*)d_in[4];
    const unsigned char* amask = (const unsigned char*)d_in[5];
    const float* Wq = (const float*)d_in[6];
    const float* Wk = (const float*)d_in[7];
    const float* Wv = (const float*)d_in[8];
    const float* Wo = (const float*)d_in[9];

    float* out        = (float*)d_out;                     // (L, NB, E)
    float* logits_out = out + (size_t)L * NB * E;          // (BH, L, S)

    // workspace (f16): 4 weights (8 MB) + q/k/v/o plain tensors (32 MB)
    _Float16* Wq16 = (_Float16*)d_ws;
    _Float16* Wk16 = Wq16 + (size_t)E * E;
    _Float16* Wv16 = Wk16 + (size_t)E * E;
    _Float16* Wo16 = Wv16 + (size_t)E * E;
    _Float16* qpr  = Wo16 + (size_t)E * E;
    _Float16* kpr  = qpr + (size_t)L * NB * E;
    _Float16* vpr  = kpr + (size_t)S * NB * E;
    _Float16* opr  = vpr + (size_t)S * NB * E;

    const int nW = E * E;
    cvt_f32_f16<<<nW / 1024, 256, 0, stream>>>(Wq, Wq16, nW);
    cvt_f32_f16<<<nW / 1024, 256, 0, stream>>>(Wk, Wk16, nW);
    cvt_f32_f16<<<nW / 1024, 256, 0, stream>>>(Wv, Wv16, nW);
    cvt_f32_f16<<<nW / 1024, 256, 0, stream>>>(Wo, Wo16, nW);

    dim3 gp(32, 16, 1);  // M=4096/128, N=1024/64
    proj_gemm<<<gp, 256, 0, stream>>>(query, Wq16, qpr);
    proj_gemm<<<gp, 256, 0, stream>>>(key,   Wk16, kpr);
    proj_gemm<<<gp, 256, 0, stream>>>(value, Wv16, vpr);

    attn_kernel<<<dim3(L / 128, BH), 256, 0, stream>>>(qpr, kpr, vpr, prev, amask, kpm,
                                                       logits_out, opr);

    out_gemm<<<dim3(32, 16), 256, 0, stream>>>(opr, Wo16, out);
}
